// WaveletScatteringTransform_81003083202984
// MI455X (gfx1250) — matrix operand order repaired and checked by reading, not compiled, not run on hardware
//
#include <hip/hip_runtime.h>
#include <hip/hip_bf16.h>

typedef __attribute__((ext_vector_type(16))) __bf16 v16bf;
typedef __attribute__((ext_vector_type(8)))  float  v8f;
typedef unsigned int u32;

#define FN      24
#define SLEN    8192
#define NB      8
#define KLEN    2048
#define PHILEN  64
#define XPAD    (SLEN + 2048)        // 10240, data offset +1024
#define U2ROW   (SLEN + 64)          // 8256,  data offset +32
#define NPAIR   276

// workspace layout (bf16 element offsets)
#define XPAD_OFF   0
#define PSIRE_OFF  (NB * XPAD)                         // 81920
#define PSIIM_OFF  (PSIRE_OFF + 32 * KLEN)             // 147456
#define U1_OFF     (PSIIM_OFF + 32 * KLEN)             // 212992
#define U2_OFF     (U1_OFF + NB * FN * XPAD)           // 2179072
#define WS_ELEMS   (U2_OFF + NB * NPAIR * U2ROW)       // 20408320
#define WS_DWORDS  (WS_ELEMS / 2)                      // 10204160

// gfx1250 async global->LDS path (ASYNCcnt-tracked DMA, no VGPR round-trip)
#if defined(__HIP_DEVICE_COMPILE__)
#  if __has_builtin(__builtin_amdgcn_global_load_async_to_lds_b128) && \
      __has_builtin(__builtin_amdgcn_s_wait_asynccnt)
#    define USE_ASYNC_LDS 1
#  else
#    define USE_ASYNC_LDS 0
#  endif
#else
#  define USE_ASYNC_LDS 0
#endif

// ---------------- K0a: zero workspace (pads must be zero) ----------------
__global__ void zero_ws_kernel(u32* ws, int ndw) {
    int i = blockIdx.x * blockDim.x + threadIdx.x;
    int stride = gridDim.x * blockDim.x;
    for (; i < ndw; i += stride) ws[i] = 0u;
}

// ---------------- K0b: f32 -> bf16 converts into padded layouts ----------
__global__ void init_ws_kernel(const float* __restrict__ sig,
                               const float* __restrict__ pre,
                               const float* __restrict__ pim,
                               __bf16* ws) {
    int idx = blockIdx.x * blockDim.x + threadIdx.x;
    if (idx < NB * SLEN) {
        int bb = idx >> 13, t = idx & (SLEN - 1);
        ws[XPAD_OFF + bb * XPAD + 1024 + t] = (__bf16)sig[idx];
    } else if (idx < NB * SLEN + FN * KLEN) {
        int k2 = idx - NB * SLEN;            // rows 0..23 contiguous; rows 24..31 stay zero
        ws[PSIRE_OFF + k2] = (__bf16)pre[k2];
        ws[PSIIM_OFF + k2] = (__bf16)pim[k2];
    }
}

// ---------------- WMMA fragment loaders ----------------
// A (16-bit 16x32): lane m = lane&15; K = kk*32 + 16*hi + 0..15
__device__ __forceinline__ v16bf load_afrag(const __bf16* row, int kk, int hi) {
    union { v16bf v; uint4 q[2]; } u;
    const uint4* p = (const uint4*)(row + (kk << 5) + (hi << 4));
    u.q[0] = p[0];
    u.q[1] = p[1];          // +8 elements = +16B
    return u.v;
}
// B (16-bit 32x16): lane col n = lane&15; K = kk*32 + 16*hi + e  -> 16 consecutive
__device__ __forceinline__ v16bf load_bfrag(const __bf16* lds, int base) {
    union { v16bf v; __bf16 h[16]; } u;
#pragma unroll
    for (int e = 0; e < 16; ++e) u.h[e] = lds[base + e];
    return u.v;
}

#define WMMA_BF16(A, Bf, C) \
    __builtin_amdgcn_wmma_f32_16x16x32_bf16(false, (A), false, (Bf), (short)0, (C), false, false)

// ---------------- shared implicit-GEMM mainloop ----------------
// block = 256 threads (8 waves); wave w computes time subtile [t0b+16w, +16)
// channel tile ct1 (ch 16..31) always, ct0 (ch 0..15) iff doT0.
// Each iteration: 2 K-chunks -> one big load clause (16x b128 + 4x ds_b128)
// followed by 8 WMMAs consuming them under descending partial waits.
__device__ __forceinline__ void gemm_tile(const __bf16* xrow,
                                          const __bf16* psiRe,
                                          const __bf16* psiIm,
                                          __bf16* lds, int t0b, bool doT0,
                                          v8f& re0, v8f& im0, v8f& re1, v8f& im1) {
    const int tid  = threadIdx.x;
    const int lane = tid & 31;
    const int wave = tid >> 5;
    const int n    = lane & 15;
    const int hi   = lane >> 4;

    // stage 2176 bf16 signal window -> LDS
#if USE_ASYNC_LDS
    {
        typedef int vint4 __attribute__((vector_size(16)));
        typedef __attribute__((address_space(1))) vint4 as1_vint4;
        typedef __attribute__((address_space(3))) vint4 as3_vint4;
        const __bf16* g = xrow + t0b;
        for (int i = tid; i < 272; i += 256) {   // 272 x 16B
            __builtin_amdgcn_global_load_async_to_lds_b128(
                (as1_vint4*)(g + i * 8),
                (as3_vint4*)(lds + i * 8),
                0, 0);
        }
        __builtin_amdgcn_s_wait_asynccnt(0);
    }
#else
    {
        __builtin_prefetch(xrow + t0b, 0, 1);    // global_prefetch_b8
        const u32* s = (const u32*)(xrow + t0b);
        u32* d = (u32*)lds;
        for (int i = tid; i < 1088; i += 256) d[i] = s[i];
    }
#endif
    __syncthreads();

    const __bf16* ar0p = psiRe + (n) * KLEN;
    const __bf16* ai0p = psiIm + (n) * KLEN;
    const __bf16* ar1p = psiRe + (16 + n) * KLEN;
    const __bf16* ai1p = psiIm + (16 + n) * KLEN;
    const int bbase = (wave << 4) + n + (hi << 4) + 1;

#pragma unroll 4
    for (int kk = 0; kk < KLEN / 32; kk += 2) {
        // --- load batch: all fragments for chunks kk and kk+1 ---
        v16bf bf0  = load_bfrag(lds, bbase + (kk << 5));
        v16bf bf1  = load_bfrag(lds, bbase + ((kk + 1) << 5));
        v16bf a10r = load_afrag(ar1p, kk, hi);
        v16bf a10i = load_afrag(ai1p, kk, hi);
        v16bf a11r = load_afrag(ar1p, kk + 1, hi);
        v16bf a11i = load_afrag(ai1p, kk + 1, hi);
        if (doT0) {   // block-uniform branch: EXEC stays all-ones around WMMA
            v16bf a00r = load_afrag(ar0p, kk, hi);
            v16bf a00i = load_afrag(ai0p, kk, hi);
            v16bf a01r = load_afrag(ar0p, kk + 1, hi);
            v16bf a01i = load_afrag(ai0p, kk + 1, hi);
            // --- 8 WMMAs, accumulator chains interleaved ---
            re1 = WMMA_BF16(a10r, bf0, re1);
            im1 = WMMA_BF16(a10i, bf0, im1);
            re0 = WMMA_BF16(a00r, bf0, re0);
            im0 = WMMA_BF16(a00i, bf0, im0);
            re1 = WMMA_BF16(a11r, bf1, re1);
            im1 = WMMA_BF16(a11i, bf1, im1);
            re0 = WMMA_BF16(a01r, bf1, re0);
            im0 = WMMA_BF16(a01i, bf1, im0);
        } else {
            re1 = WMMA_BF16(a10r, bf0, re1);
            im1 = WMMA_BF16(a10i, bf0, im1);
            re1 = WMMA_BF16(a11r, bf1, re1);
            im1 = WMMA_BF16(a11i, bf1, im1);
        }
    }
}

// ---------------- K1: u1 = |x * psi| ----------------
__global__ void __launch_bounds__(256) order1_kernel(__bf16* ws) {
    __shared__ __attribute__((aligned(16))) __bf16 xs[2176];
    const int t0b = blockIdx.x << 7;     // 64 blocks * 128 t
    const int bb  = blockIdx.y;          // batch

    v8f re0 = {}, im0 = {}, re1 = {}, im1 = {};
    gemm_tile(ws + XPAD_OFF + bb * XPAD, ws + PSIRE_OFF, ws + PSIIM_OFF,
              xs, t0b, true, re0, im0, re1, im1);

    const int lane = threadIdx.x & 31, wave = threadIdx.x >> 5;
    const int n = lane & 15, hi = lane >> 4;
    const int t = t0b + (wave << 4) + n;
    __bf16* u1 = ws + U1_OFF;
#pragma unroll
    for (int r = 0; r < 8; ++r) {
        int ch0 = r + (hi << 3);                  // rows: M = vgpr + 8*hi
        float v0 = __builtin_sqrtf(re0[r] * re0[r] + im0[r] * im0[r]);
        u1[(bb * FN + ch0) * XPAD + 1024 + t] = (__bf16)v0;
        int ch1 = 16 + ch0;
        if (ch1 < FN) {
            float v1 = __builtin_sqrtf(re1[r] * re1[r] + im1[r] * im1[r]);
            u1[(bb * FN + ch1) * XPAD + 1024 + t] = (__bf16)v1;
        }
    }
}

// ---------------- K2: u2(i,j) = |u1_i * psi_j|, only j > i ----------------
__global__ void __launch_bounds__(256) order2_kernel(__bf16* ws) {
    __shared__ __attribute__((aligned(16))) __bf16 xs[2176];
    const int t0b = blockIdx.x << 7;
    const int bi  = blockIdx.y;               // 0..191
    const int bb  = bi / FN, i = bi % FN;
    if (i == FN - 1) return;                  // no j > 23 (uniform exit)
    const bool doT0 = (i < 15);               // tile j=0..15 only useful if i<15

    v8f re0 = {}, im0 = {}, re1 = {}, im1 = {};
    gemm_tile(ws + U1_OFF + (bb * FN + i) * XPAD, ws + PSIRE_OFF, ws + PSIIM_OFF,
              xs, t0b, doT0, re0, im0, re1, im1);

    const int lane = threadIdx.x & 31, wave = threadIdx.x >> 5;
    const int n = lane & 15, hi = lane >> 4;
    const int t = t0b + (wave << 4) + n;
    __bf16* u2 = ws + U2_OFF;
    const int pbase = 23 * i - (i * (i - 1)) / 2;   // triu(k=1) row start
#pragma unroll
    for (int r = 0; r < 8; ++r) {
        int j0 = r + (hi << 3);
        if (doT0 && j0 > i) {
            float v = __builtin_sqrtf(re0[r] * re0[r] + im0[r] * im0[r]);
            int p = pbase + (j0 - i - 1);
            u2[(bb * NPAIR + p) * U2ROW + 32 + t] = (__bf16)v;
        }
        int j1 = 16 + j0;
        if (j1 > i && j1 < FN) {
            float v = __builtin_sqrtf(re1[r] * re1[r] + im1[r] * im1[r]);
            int p = pbase + (j1 - i - 1);
            u2[(bb * NPAIR + p) * U2ROW + 32 + t] = (__bf16)v;
        }
    }
}

// ---------------- K3: Gaussian lowpass phi -> final output ----------------
__global__ void __launch_bounds__(256) lowpass_kernel(const __bf16* __restrict__ ws,
                                                      const float* __restrict__ phi,
                                                      float* __restrict__ out) {
    int idx  = blockIdx.x * 256 + threadIdx.x;     // 8*300*8192 total
    int t    = idx & (SLEN - 1);
    int rest = idx >> 13;
    int ch   = rest % 300;                          // uniform per block
    int bb   = rest / 300;

    const __bf16* src; int base, phiRow;
    if (ch < FN) {                                  // s1: phi[ch]
        src = ws + U1_OFF + (bb * FN + ch) * XPAD;
        base = 1024 + t - 31; phiRow = ch;
    } else {                                        // s2 pair p -> phi[i]
        int p = ch - FN, rem = p, i = 0;
        while (rem >= FN - 1 - i) { rem -= FN - 1 - i; ++i; }
        src = ws + U2_OFF + (bb * NPAIR + p) * U2ROW;
        base = 32 + t - 31; phiRow = i;
    }
    const float* ph = phi + phiRow * PHILEN;
    float s = 0.f;
#pragma unroll
    for (int k = 0; k < PHILEN; ++k) s += (float)src[base + k] * ph[k];
    out[idx] = s;
}

// ---------------- launcher ----------------
extern "C" void kernel_launch(void* const* d_in, const int* in_sizes, int n_in,
                              void* d_out, int out_size, void* d_ws, size_t ws_size,
                              hipStream_t stream) {
    (void)in_sizes; (void)n_in; (void)out_size; (void)ws_size;
    const float* sig = (const float*)d_in[0];
    const float* pre = (const float*)d_in[1];
    const float* pim = (const float*)d_in[2];
    const float* phi = (const float*)d_in[3];
    __bf16* ws = (__bf16*)d_ws;
    float* out = (float*)d_out;

    zero_ws_kernel<<<2048, 256, 0, stream>>>((u32*)d_ws, WS_DWORDS);
    init_ws_kernel<<<(NB * SLEN + FN * KLEN) / 256, 256, 0, stream>>>(sig, pre, pim, ws);
    order1_kernel<<<dim3(SLEN / 128, NB), 256, 0, stream>>>(ws);
    order2_kernel<<<dim3(SLEN / 128, NB * FN), 256, 0, stream>>>(ws);
    lowpass_kernel<<<(NB * 300 * SLEN) / 256, 256, 0, stream>>>(ws, phi, out);
}
